// PeriodicKernel_12438225289613
// MI455X (gfx1250) — matrix operand order repaired and checked by reading, not compiled, not run on hardware
//
#include <hip/hip_runtime.h>
#include <hip/hip_bf16.h>

// ---------------------------------------------------------------------------
// Periodic kernel: out = exp(-2*sin(pi*dist/p)^2 / ls[col]^2)
// dist from cdist(x1/ls, x2/ls) via ||a||^2+||b||^2-2ab (GEMM on WMMA bf16x3)
// ---------------------------------------------------------------------------

typedef __attribute__((ext_vector_type(8)))  float  v8f;
typedef __attribute__((ext_vector_type(16))) __bf16 v16bf;
typedef __attribute__((ext_vector_type(8)))  __bf16 v8bf;
typedef __attribute__((ext_vector_type(4)))  int    v4i;

union Frag16 { v16bf v; v8bf h[2]; };

#define BM 128
#define BN 128
#define KS 32
#define PI_F 3.1415926f

#if defined(__HIP_DEVICE_COMPILE__) && \
    __has_builtin(__builtin_amdgcn_global_load_async_to_lds_b128) && \
    __has_builtin(__builtin_amdgcn_s_wait_asynccnt)
#define HAVE_ASYNC_LDS 1
#else
#define HAVE_ASYNC_LDS 0
#endif

#if HAVE_ASYNC_LDS
typedef __attribute__((address_space(1))) v4i GV4;  // global int4
typedef __attribute__((address_space(3))) v4i LV4;  // LDS int4
#endif

// 16-byte global -> LDS copy (async-to-LDS on CDNA5 when the builtin exists)
__device__ __forceinline__ void copy16_g2l(__bf16* lds_dst, const __bf16* gsrc) {
#if HAVE_ASYNC_LDS
  __builtin_amdgcn_global_load_async_to_lds_b128(
      (GV4*)(unsigned long long)(const void*)gsrc,
      (LV4*)(unsigned int)(unsigned long long)(const void*)lds_dst,
      0, 0);
#else
  *(v8bf*)lds_dst = *(const v8bf*)gsrc;
#endif
}

template <int N> __device__ __forceinline__ void wait_async_le() {
#if HAVE_ASYNC_LDS
  __builtin_amdgcn_s_wait_asynccnt(N);
#endif
}

// ---------------------------------------------------------------------------
// Kernel 1: per-dimension prep:
//   inv_ls[j]   = exp(-log_ls[j])            (so s = x * inv_ls)
//   colfact[j]  = -2 * exp(-2*log_ls[j])     (epilogue:  exp(colfact*sin^2))
//   inv_p[0]    = exp(-log_p)
// ---------------------------------------------------------------------------
__global__ void prep_kernel(const float* __restrict__ log_ls,
                            const float* __restrict__ log_p,
                            float* __restrict__ inv_ls,
                            float* __restrict__ colfact,
                            float* __restrict__ inv_p, int d) {
  int j = blockIdx.x * blockDim.x + threadIdx.x;
  if (j < d) {
    float L = log_ls[j];
    inv_ls[j]  = __expf(-L);
    colfact[j] = -2.0f * __expf(-2.0f * L);
  }
  if (j == 0) inv_p[0] = __expf(-log_p[0]);
}

// ---------------------------------------------------------------------------
// Kernel 2: scale rows by inv_ls, split into bf16 hi/lo, and row sq-norms.
// One block (256 threads) per row.
// ---------------------------------------------------------------------------
__global__ __launch_bounds__(256)
void scale_split_kernel(const float* __restrict__ X,
                        const float* __restrict__ inv_ls,
                        __bf16* __restrict__ Hi, __bf16* __restrict__ Lo,
                        float* __restrict__ sq, int d) {
  const int row = blockIdx.x;
  const int t   = threadIdx.x;
  const float* xr = X + (size_t)row * d;
  __bf16* hr = Hi + (size_t)row * d;
  __bf16* lr = Lo + (size_t)row * d;

  float sum = 0.0f;
  for (int k = t; k < d; k += 256) {
    float s = xr[k] * inv_ls[k];
    __bf16 h = (__bf16)s;
    float  l = s - (float)h;
    hr[k] = h;
    lr[k] = (__bf16)l;
    sum += s * s;
  }
  // wave32 shuffle reduce, then cross-wave via LDS
  #pragma unroll
  for (int off = 16; off > 0; off >>= 1) sum += __shfl_down(sum, off, 32);
  __shared__ float red[8];
  if ((t & 31) == 0) red[t >> 5] = sum;
  __syncthreads();
  if (t == 0) {
    float tot = 0.0f;
    #pragma unroll
    for (int i = 0; i < 8; ++i) tot += red[i];
    sq[row] = tot;
  }
}

// ---------------------------------------------------------------------------
// Kernel 3: WMMA GEMM (cross = S1 * S2^T, bf16 hi/lo x3 emulation of f32)
// fused with the periodic-kernel epilogue.
//   Block tile 128x128, 8 wave32 (2x4), wave tile 64x32 => 4x2 accum tiles.
//   K staged through LDS in chunks of 32, double buffered (64KB LDS).
// ---------------------------------------------------------------------------
__global__ __launch_bounds__(256)
void periodic_wmma_gemm(const __bf16* __restrict__ Ah, const __bf16* __restrict__ Al,
                        const __bf16* __restrict__ Bh, const __bf16* __restrict__ Bl,
                        const float* __restrict__ sq1, const float* __restrict__ sq2,
                        const float* __restrict__ colfact, const float* __restrict__ inv_p_ws,
                        float* __restrict__ out, int R, int C, int D) {
  __shared__ __bf16 sA[2][2][BM][KS];  // [buf][hi/lo][row][k]   32 KB
  __shared__ __bf16 sB[2][2][BN][KS];  // [buf][hi/lo][col][k]   32 KB

  const int t    = threadIdx.x;
  const int lane = t & 31;
  const int wave = t >> 5;
  const int wr   = wave >> 2;   // 0..1  (wave row: 64 rows each)
  const int wc   = wave & 3;    // 0..3  (wave col: 32 cols each)
  const int rowBase = blockIdx.y * BM;
  const int colBase = blockIdx.x * BN;
  if (rowBase >= R || colBase >= C) return;

  // cooperative load mapping: thread t covers 16 contiguous bf16 of a tile row
  const int lrow = t >> 1;          // 0..127
  const int lko  = (t & 1) * 16;    // 0 or 16

  auto load_stage = [&](int buf, int k0) {
    const __bf16* gah = Ah + (size_t)(rowBase + lrow) * D + k0 + lko;
    const __bf16* gal = Al + (size_t)(rowBase + lrow) * D + k0 + lko;
    const __bf16* gbh = Bh + (size_t)(colBase + lrow) * D + k0 + lko;
    const __bf16* gbl = Bl + (size_t)(colBase + lrow) * D + k0 + lko;
    copy16_g2l(&sA[buf][0][lrow][lko],     gah);
    copy16_g2l(&sA[buf][0][lrow][lko + 8], gah + 8);
    copy16_g2l(&sA[buf][1][lrow][lko],     gal);
    copy16_g2l(&sA[buf][1][lrow][lko + 8], gal + 8);
    copy16_g2l(&sB[buf][0][lrow][lko],     gbh);
    copy16_g2l(&sB[buf][0][lrow][lko + 8], gbh + 8);
    copy16_g2l(&sB[buf][1][lrow][lko],     gbl);
    copy16_g2l(&sB[buf][1][lrow][lko + 8], gbl + 8);
  };

  v8f acc[4][2];
  #pragma unroll
  for (int mt = 0; mt < 4; ++mt)
    #pragma unroll
    for (int nt = 0; nt < 2; ++nt) acc[mt][nt] = (v8f)0.0f;

  const int lm = lane & 15;
  const int ka = (lane < 16) ? 0 : 8;    // 16-bit A layout: K chunks of 8 per half-wave
  const int kb = (lane < 16) ? 0 : 8;    // 16-bit B layout: K chunks of 8 per half-wave

  auto compute_stage = [&](int buf) {
    Frag16 fah[4], fal[4], fbh[2], fbl[2];
    #pragma unroll
    for (int mt = 0; mt < 4; ++mt) {
      const int rr = wr * 64 + mt * 16 + lm;
      fah[mt].h[0] = *(const v8bf*)&sA[buf][0][rr][ka];
      fah[mt].h[1] = *(const v8bf*)&sA[buf][0][rr][ka + 16];
      fal[mt].h[0] = *(const v8bf*)&sA[buf][1][rr][ka];
      fal[mt].h[1] = *(const v8bf*)&sA[buf][1][rr][ka + 16];
    }
    #pragma unroll
    for (int nt = 0; nt < 2; ++nt) {
      const int cc = wc * 32 + nt * 16 + lm;
      fbh[nt].h[0] = *(const v8bf*)&sB[buf][0][cc][kb];
      fbh[nt].h[1] = *(const v8bf*)&sB[buf][0][cc][kb + 16];
      fbl[nt].h[0] = *(const v8bf*)&sB[buf][1][cc][kb];
      fbl[nt].h[1] = *(const v8bf*)&sB[buf][1][cc][kb + 16];
    }
    #pragma unroll
    for (int mt = 0; mt < 4; ++mt) {
      #pragma unroll
      for (int nt = 0; nt < 2; ++nt) {
        acc[mt][nt] = __builtin_amdgcn_wmma_f32_16x16x32_bf16(
            false, fah[mt].v, false, fbh[nt].v, (short)0, acc[mt][nt], false, false);
        acc[mt][nt] = __builtin_amdgcn_wmma_f32_16x16x32_bf16(
            false, fah[mt].v, false, fbl[nt].v, (short)0, acc[mt][nt], false, false);
        acc[mt][nt] = __builtin_amdgcn_wmma_f32_16x16x32_bf16(
            false, fal[mt].v, false, fbh[nt].v, (short)0, acc[mt][nt], false, false);
      }
    }
  };

  const int NK = D / KS;
  load_stage(0, 0);
  for (int ks = 0; ks < NK; ++ks) {
    if (ks + 1 < NK) {
      load_stage((ks + 1) & 1, (ks + 1) * KS);
      wait_async_le<8>();   // stage `ks` fully landed; next-stage 8 copies in flight
    } else {
      wait_async_le<0>();
    }
    __syncthreads();
    compute_stage(ks & 1);
    __syncthreads();
  }

  // ---- epilogue: dist -> periodic kernel ----
  const float invp = inv_p_ws[0];
  #pragma unroll
  for (int mt = 0; mt < 4; ++mt) {
    const int rbase = rowBase + wr * 64 + mt * 16 + ((lane >> 4) << 3);
    float s1v[8];
    #pragma unroll
    for (int r = 0; r < 8; ++r) s1v[r] = sq1[rbase + r];
    #pragma unroll
    for (int nt = 0; nt < 2; ++nt) {
      const int col = colBase + wc * 32 + nt * 16 + lm;
      const float s2v = sq2[col];
      const float cf  = colfact[col];
      #pragma unroll
      for (int r = 0; r < 8; ++r) {
        float cross = acc[mt][nt][r];
        float d2    = fmaxf(s1v[r] + s2v - 2.0f * cross, 0.0f);
        float dist  = __builtin_sqrtf(d2);
        float sn    = __sinf(PI_F * dist * invp);
        out[(size_t)(rbase + r) * C + col] = __expf(cf * sn * sn);
      }
    }
  }
}

// ---------------------------------------------------------------------------
// Host launcher. Workspace layout (d=2048, n=8192, m=2048 => ~80.2 MB):
//   Ah,Al: n*d bf16 each; Bh,Bl: m*d bf16 each; sq1[n], sq2[m],
//   inv_ls[d], colfact[d], inv_p (all f32).
// ---------------------------------------------------------------------------
extern "C" void kernel_launch(void* const* d_in, const int* in_sizes, int n_in,
                              void* d_out, int out_size, void* d_ws, size_t ws_size,
                              hipStream_t stream) {
  (void)n_in; (void)out_size; (void)ws_size;
  const float* x1     = (const float*)d_in[0];
  const float* x2     = (const float*)d_in[1];
  const float* log_ls = (const float*)d_in[2];
  const float* log_p  = (const float*)d_in[3];
  float* out = (float*)d_out;

  const int d = in_sizes[2];
  const int n = in_sizes[0] / d;
  const int m = in_sizes[1] / d;

  char* w = (char*)d_ws;
  auto take = [&](size_t bytes) {
    char* p = w;
    w += (bytes + 255) & ~(size_t)255;
    return p;
  };
  __bf16* Ah = (__bf16*)take((size_t)n * d * sizeof(__bf16));
  __bf16* Al = (__bf16*)take((size_t)n * d * sizeof(__bf16));
  __bf16* Bh = (__bf16*)take((size_t)m * d * sizeof(__bf16));
  __bf16* Bl = (__bf16*)take((size_t)m * d * sizeof(__bf16));
  float* sq1     = (float*)take((size_t)n * sizeof(float));
  float* sq2     = (float*)take((size_t)m * sizeof(float));
  float* inv_ls  = (float*)take((size_t)d * sizeof(float));
  float* colfact = (float*)take((size_t)d * sizeof(float));
  float* inv_p   = (float*)take(256);

  prep_kernel<<<(d + 255) / 256, 256, 0, stream>>>(log_ls, log_p, inv_ls, colfact, inv_p, d);
  scale_split_kernel<<<n, 256, 0, stream>>>(x1, inv_ls, Ah, Al, sq1, d);
  scale_split_kernel<<<m, 256, 0, stream>>>(x2, inv_ls, Bh, Bl, sq2, d);

  dim3 grid(m / BN, n / BM);
  periodic_wmma_gemm<<<grid, 256, 0, stream>>>(Ah, Al, Bh, Bl, sq1, sq2, colfact, inv_p,
                                               out, n, m, d);
}
